// MultiLayerRNN_16844861735434
// MI455X (gfx1250) — matrix operand order repaired and checked by reading, not compiled, not run on hardware
//
#include <hip/hip_runtime.h>
#include <hip/hip_bf16.h>

// ---------------------------------------------------------------------------
// 2-layer tanh RNN, B=64 T=512 I=H=512, persistent WMMA-bf16 kernel.
// Weights staged once in LDS via gfx1250 async global->LDS copies.
// Inner loop: batched operand preloads (immediate offsets) + dual WMMA chains.
// ---------------------------------------------------------------------------

typedef __attribute__((ext_vector_type(16))) __bf16 v16bf;
typedef __attribute__((ext_vector_type(8)))  __bf16 v8bf;
typedef __attribute__((ext_vector_type(8)))  float  v8f;

#define BB 64
#define TT 512
#define II 512
#define HH 512
#define KK 1024              // I + H concatenated reduction dim
#define NGRP 256             // persistent grid size (2 layers x 128 tiles)
#define WPAD 1032            // LDS weight row stride (1024 + 8 bf16 pad)

// ---- workspace layout (bytes) ---------------------------------------------
#define WS_CNT   0                                   // 256 B (sync counter)
#define WS_XT    256                                 // [T][B][I] bf16 = 32 MB
#define WS_W     (WS_XT + (size_t)TT*BB*II*2)        // [L][H][K] bf16 = 2 MB
#define WS_BIAS  (WS_W + (size_t)2*HH*KK*2)          // [L][H] f32 = 4 KB
#define WS_HBUF  (WS_BIAS + (size_t)2*HH*4)          // 4 x [B][H] bf16 = 256 KB
// hbuf(l,p) element offset = (l*2+p)*BB*HH

// ---------------------------------------------------------------------------
// prep kernels
// ---------------------------------------------------------------------------
__global__ void k_prep_x(const float* __restrict__ x, __bf16* __restrict__ xT) {
  size_t idx = (size_t)blockIdx.x * 256 + threadIdx.x;   // B*T*I threads
  int b = (int)(idx / ((size_t)TT * II));
  int r = (int)(idx % ((size_t)TT * II));
  int t = r / II;
  int i = r % II;
  xT[((size_t)t * BB + b) * II + i] = (__bf16)x[idx];
}

__global__ void k_prep_w(const float* __restrict__ Wih, const float* __restrict__ Whh,
                         __bf16* __restrict__ W) {
  int idx = blockIdx.x * 256 + threadIdx.x;              // 2*H*K threads
  int l = idx >> 19;                                     // H*K = 2^19
  int r = idx & ((1 << 19) - 1);
  int n = r >> 10;                                       // K = 1024
  int k = r & (KK - 1);
  float v = (k < II) ? Wih[((size_t)(l * HH + n)) * II + k]
                     : Whh[((size_t)(l * HH + n)) * HH + (k - II)];
  W[idx] = (__bf16)v;
}

__global__ void k_prep_misc(const float* __restrict__ h0,
                            const float* __restrict__ bih, const float* __restrict__ bhh,
                            float* __restrict__ bias, __bf16* __restrict__ hb,
                            unsigned* __restrict__ cnt) {
  int idx = blockIdx.x * 256 + threadIdx.x;              // 2*B*H threads
  if (idx < 2 * BB * HH) {
    int l = idx >> 15;                                   // B*H = 32768
    int r = idx & (BB * HH - 1);
    hb[(size_t)(l * 2 + 1) * (BB * HH) + r] = (__bf16)h0[idx];   // parity of t=-1
  }
  if (idx < 2 * HH) bias[idx] = bih[idx] + bhh[idx];
  if (idx == 0) *cnt = 0u;
}

// ---------------------------------------------------------------------------
__device__ __forceinline__ v16bf cat16(v8bf lo, v8bf hi) {
  return __builtin_shufflevector(lo, hi, 0,1,2,3,4,5,6,7,8,9,10,11,12,13,14,15);
}

// ---------------------------------------------------------------------------
// device-wide barrier: monotonic counter, acq_rel at agent scope
// ---------------------------------------------------------------------------
__device__ __forceinline__ void grid_barrier(unsigned* cnt, unsigned it) {
  __threadfence();
  __syncthreads();
  if (threadIdx.x == 0) {
    __hip_atomic_fetch_add(cnt, 1u, __ATOMIC_ACQ_REL, __HIP_MEMORY_SCOPE_AGENT);
    const unsigned target = (it + 1u) * (unsigned)NGRP;
    while (__hip_atomic_load(cnt, __ATOMIC_ACQUIRE, __HIP_MEMORY_SCOPE_AGENT) < target) {
      __builtin_amdgcn_s_sleep(1);
    }
  }
  __syncthreads();
  __threadfence();
}

// ---------------------------------------------------------------------------
// persistent RNN kernel.
//   wg < 128  : layer0 tile (t = it)
//   wg >= 128 : layer1 tile (t = it-1)   [pipelined: independent work]
// Each WG: one 16x16 output tile; 8 waves split K=1024 (128 each, 4 WMMAs),
// weights LDS-resident (async-copied once), distributed LDS reduce + tanh.
// ---------------------------------------------------------------------------
__global__ void __launch_bounds__(256)
k_rnn_persistent(const __bf16* __restrict__ xT, const __bf16* __restrict__ Wcat,
                 const float* __restrict__ bias, __bf16* __restrict__ hbase,
                 float* __restrict__ out, unsigned* __restrict__ cnt) {
  const int wg    = blockIdx.x;
  const int layer = wg >> 7;
  const int tile  = wg & 127;
  const int m0    = (tile >> 5) << 4;   // batch tile: 0..3 -> rows 0..63
  const int n0    = (tile & 31) << 4;   // hidden tile: 0..31 -> cols 0..511
  const int lane  = threadIdx.x & 31;
  const int wave  = threadIdx.x >> 5;

  __shared__ __align__(16) __bf16 wlds[16 * WPAD];   // 33 KB weight slice
  __shared__ float red[8 * 256];                     // red[r][wave][lane]

  // ---- stage this WG's weight slice (rows n0..n0+15, K=0..1023) into LDS ----
  {
    const __bf16* Wg = Wcat + (size_t)layer * HH * KK + (size_t)n0 * KK;
#pragma unroll
    for (int i = 0; i < 8; ++i) {
      const int c    = (int)threadIdx.x + i * 256;   // 0..2047 16-byte chunks
      const int row  = c >> 7;
      const int col8 = c & 127;
      const __bf16* g = Wg + (size_t)row * KK + col8 * 8;
      const unsigned loff = (unsigned)(size_t)(&wlds[row * WPAD + col8 * 8]);
      asm volatile("global_load_async_to_lds_b128 %0, %1, off"
                   :: "v"(loff), "v"(g) : "memory");
    }
    asm volatile("s_wait_asynccnt 0x0" ::: "memory");
    __syncthreads();
  }

  const float bn = bias[layer * HH + n0 + (lane & 15)];

  // per-lane invariant offsets (elements)
  //   A tile (16x32): lanes 0-15 rows m0..m0+15 k={0..7,16..23}; lanes 16-31 k+=8
  const int aoff = (m0 + (lane & 15)) * HH + ((lane >> 4) << 3);
  //   B tile (32x16): lanes 0-15 N-row, K={0..7,16..23}; lanes 16-31 k+=8
  const __bf16* bp = wlds + (lane & 15) * WPAD + ((lane >> 4) << 3) + wave * 128;

  for (int it = 0; it <= TT; ++it) {
    const int t = (layer == 0) ? it : (it - 1);
    const bool active = (t >= 0) && (t < TT);
    if (active) {
      const int p = t & 1;
      const __bf16* A1;
      const __bf16* A2;
      __bf16* hout;
      if (layer == 0) {
        A1   = xT + (size_t)t * (BB * II);                       // x_t  (K 0..511)
        A2   = hbase + (size_t)(0 * 2 + (p ^ 1)) * (BB * HH);    // h0[t-1]
        hout = hbase + (size_t)(0 * 2 + p) * (BB * HH);          // h0[t]
      } else {
        A1   = hbase + (size_t)(0 * 2 + p) * (BB * HH);          // h0[t]
        A2   = hbase + (size_t)(1 * 2 + (p ^ 1)) * (BB * HH);    // h1[t-1]
        hout = hbase + (size_t)(1 * 2 + p) * (BB * HH);          // h1[t]
      }

      // wave-uniform source select (waves 0-3 read A1, waves 4-7 read A2)
      const __bf16* ap = ((wave < 4) ? (A1 + wave * 128)
                                     : (A2 + (wave * 128 - II))) + aoff;

      // ---- batched operand preload: 8 global + 8 DS b128 loads, imm offsets
      v8bf alo[4], ahi[4], blo[4], bhi[4];
#pragma unroll
      for (int j = 0; j < 4; ++j) {
        alo[j] = *(const v8bf*)(ap + j * 32);
        ahi[j] = *(const v8bf*)(ap + j * 32 + 16);
      }
#pragma unroll
      for (int j = 0; j < 4; ++j) {
        blo[j] = *(const v8bf*)(bp + j * 32);
        bhi[j] = *(const v8bf*)(bp + j * 32 + 16);
      }

      // ---- dual accumulator chains (hide WMMA->WMMA RAW hazard)
      v8f c0 = {0.f, 0.f, 0.f, 0.f, 0.f, 0.f, 0.f, 0.f};
      v8f c1 = {0.f, 0.f, 0.f, 0.f, 0.f, 0.f, 0.f, 0.f};
      c0 = __builtin_amdgcn_wmma_f32_16x16x32_bf16(
              false, cat16(alo[0], ahi[0]), false, cat16(blo[0], bhi[0]),
              (short)0, c0, false, false);
      c1 = __builtin_amdgcn_wmma_f32_16x16x32_bf16(
              false, cat16(alo[1], ahi[1]), false, cat16(blo[1], bhi[1]),
              (short)0, c1, false, false);
      c0 = __builtin_amdgcn_wmma_f32_16x16x32_bf16(
              false, cat16(alo[2], ahi[2]), false, cat16(blo[2], bhi[2]),
              (short)0, c0, false, false);
      c1 = __builtin_amdgcn_wmma_f32_16x16x32_bf16(
              false, cat16(alo[3], ahi[3]), false, cat16(blo[3], bhi[3]),
              (short)0, c1, false, false);

      // cross-wave K reduction: red[r][wave][lane] (conflict-free)
#pragma unroll
      for (int r = 0; r < 8; ++r) red[r * 256 + wave * 32 + lane] = c0[r] + c1[r];
      __syncthreads();

      // distributed epilogue: wave w finishes accumulator row r = w
      {
        float s = 0.f;
#pragma unroll
        for (int w = 0; w < 8; ++w) s += red[wave * 256 + w * 32 + lane];
        const int n = n0 + (lane & 15);
        const int m = m0 + ((lane >> 4) << 3) + wave;            // batch index
        const float v = tanhf(s + bn);
        hout[(size_t)m * HH + n] = (__bf16)v;
        if (layer == 1)                                          // streamed, write-once
          __builtin_nontemporal_store(v, &out[(size_t)m * (TT * HH) + (size_t)t * HH + n]);
        if (t == TT - 1)
          out[(size_t)BB * TT * HH + (size_t)(layer * BB + m) * HH + n] = v; // h_final
      }
      __syncthreads();
    }
    grid_barrier(cnt, (unsigned)it);
  }
}

// ---------------------------------------------------------------------------
extern "C" void kernel_launch(void* const* d_in, const int* in_sizes, int n_in,
                              void* d_out, int out_size, void* d_ws, size_t ws_size,
                              hipStream_t stream) {
  const float* x   = (const float*)d_in[0];   // [B,T,I]
  const float* h0  = (const float*)d_in[1];   // [L,B,H]
  const float* Wih = (const float*)d_in[2];   // [L,H,I]
  const float* bih = (const float*)d_in[3];   // [L,H]
  const float* Whh = (const float*)d_in[4];   // [L,H,H]
  const float* bhh = (const float*)d_in[5];   // [L,H]
  float* out = (float*)d_out;                 // [B,T,H] ++ [L,B,H]

  char* ws = (char*)d_ws;
  unsigned* cnt  = (unsigned*)(ws + WS_CNT);
  __bf16*  xT    = (__bf16*)(ws + WS_XT);
  __bf16*  Wc    = (__bf16*)(ws + WS_W);
  float*   bias  = (float*)(ws + WS_BIAS);
  __bf16*  hbuf  = (__bf16*)(ws + WS_HBUF);

  // prep: x transpose+bf16, weight concat+bf16, bias sum + h0 init + counter=0
  k_prep_x<<<(BB * TT * II) / 256, 256, 0, stream>>>(x, xT);
  k_prep_w<<<(2 * HH * KK) / 256, 256, 0, stream>>>(Wih, Whh, Wc);
  k_prep_misc<<<(2 * BB * HH) / 256, 256, 0, stream>>>(h0, bih, bhh, bias, hbuf, cnt);

  // persistent pipelined RNN: 256 WGs (128 per layer), 513 grid barriers
  k_rnn_persistent<<<NGRP, 256, 0, stream>>>(xT, Wc, bias, hbuf, out, cnt);
}
